// PPFTrans_28905129902239
// MI455X (gfx1250) — matrix operand order repaired and checked by reading, not compiled, not run on hardware
//
#include <hip/hip_runtime.h>
#include <hip/hip_bf16.h>
#include <math.h>

// ---------------------------------------------------------------------------
// Types for CDNA5 WMMA (wave32): v_wmma_f32_16x16x32_f16
// ---------------------------------------------------------------------------
typedef __attribute__((ext_vector_type(16))) _Float16 v16h;
typedef __attribute__((ext_vector_type(8)))  _Float16 v8h;
typedef __attribute__((ext_vector_type(8)))  float    v8f;

#define NTOK 2048
#define DMODEL 256

__device__ inline v16h mkv16(v8h lo, v8h hi) {
  v16h r;
#pragma unroll
  for (int i = 0; i < 8; i++) { r[i] = lo[i]; r[i + 8] = hi[i]; }
  return r;
}

__device__ inline v8f wmma16(v16h a, v16h b, v8f c) {
  // (neg_a, A, neg_b, B, c_mod, C, reuse_a, reuse_b)
  return __builtin_amdgcn_wmma_f32_16x16x32_f16(false, a, false, b, (short)0, c,
                                                false, false);
}

__device__ inline void atomicMaxF(float* addr, float v) {
  int* ai = (int*)addr;
  int old = *ai;
  while (__int_as_float(old) < v) {
    int assumed = old;
    old = atomicCAS(ai, assumed, __float_as_int(v));
    if (old == assumed) break;
  }
}

// ---------------------------------------------------------------------------
// Elementwise / pack kernels
// ---------------------------------------------------------------------------
__global__ void k_f32_to_f16(_Float16* __restrict__ dst,
                             const float* __restrict__ src, int n) {
  int i = blockIdx.x * 256 + threadIdx.x;
  if (i < n) dst[i] = (_Float16)src[i];
}

// src [N,D] row-major -> dst [D,N]
__global__ void k_transpose_in(float* __restrict__ dst,
                               const float* __restrict__ src) {
  int i = blockIdx.x * 256 + threadIdx.x;   // 524288
  int n = i >> 8, d = i & 255;
  dst[(size_t)d * NTOK + n] = src[i];
}

// src [D,N] -> dst [N,D]
__global__ void k_transpose_out(float* __restrict__ dst,
                                const float* __restrict__ src) {
  int i = blockIdx.x * 256 + threadIdx.x;
  int n = i >> 8, d = i & 255;
  dst[i] = src[(size_t)d * NTOK + n];
}

__global__ void k_add_inplace(float* __restrict__ a,
                              const float* __restrict__ b, int n) {
  int i = blockIdx.x * 256 + threadIdx.x;
  if (i < n) a[i] += b[i];
}

// q,k,v f32 [256][2048] -> per-head f16 layouts.
// head h = d & 3, hd = d >> 2 (reshape [HEAD=64, H=4, N]).
__global__ void k_qkv_pack(const float* __restrict__ q,
                           const float* __restrict__ k,
                           const float* __restrict__ v,
                           _Float16* __restrict__ qT,   // [h][n][hd]
                           _Float16* __restrict__ kp,   // [h][hd][n]
                           _Float16* __restrict__ vT) { // [h][m][hd]
  int i = blockIdx.x * 256 + threadIdx.x;   // 524288
  int d = i >> 11, n = i & 2047;
  int h = d & 3, hd = d >> 2;
  size_t hb = (size_t)h * NTOK * 64;
  qT[hb + (size_t)n * 64 + hd] = (_Float16)q[i];
  kp[hb + (size_t)hd * NTOK + n] = (_Float16)k[i];
  vT[hb + (size_t)n * 64 + hd] = (_Float16)v[i];
}

// attn f32 [4][2048][64] -> x16 [256][2048] with d = hd*4 + h
__global__ void k_attn_pack(const float* __restrict__ attn,
                            _Float16* __restrict__ x16) {
  int i = blockIdx.x * 256 + threadIdx.x;   // 524288 (= d*2048 + n)
  int d = i >> 11, n = i & 2047;
  int h = d & 3, hd = d >> 2;
  x16[i] = (_Float16)attn[(size_t)h * NTOK * 64 + (size_t)n * 64 + hd];
}

// row-wise softmax of scores (f32) with scale, writing f16 probabilities
__global__ __launch_bounds__(256)
void k_softmax(const float* __restrict__ S, _Float16* __restrict__ P,
               int cols, float scale) {
  int row = blockIdx.x;
  const float* x = S + (size_t)row * cols;
  _Float16* p = P + (size_t)row * cols;
  __shared__ float red[8];
  int tid = threadIdx.x, lane = tid & 31, wave = tid >> 5;
  float mx = -3.0e38f;
  for (int c = tid; c < cols; c += 256) mx = fmaxf(mx, x[c]);
  for (int o = 16; o > 0; o >>= 1) mx = fmaxf(mx, __shfl_xor(mx, o, 32));
  if (lane == 0) red[wave] = mx;
  __syncthreads();
  mx = red[0];
  for (int i = 1; i < 8; i++) mx = fmaxf(mx, red[i]);
  __syncthreads();
  float sum = 0.f;
  for (int c = tid; c < cols; c += 256) sum += __expf((x[c] - mx) * scale);
  for (int o = 16; o > 0; o >>= 1) sum += __shfl_xor(sum, o, 32);
  if (lane == 0) red[wave] = sum;
  __syncthreads();
  sum = red[0];
  for (int i = 1; i < 8; i++) sum += red[i];
  float inv = 1.f / sum;
  for (int c = tid; c < cols; c += 256)
    p[c] = (_Float16)(__expf((x[c] - mx) * scale) * inv);
}

// per-row InstanceNorm (population var) + ReLU, f32 in -> f16 out
__global__ __launch_bounds__(256)
void k_inorm_relu(const float* __restrict__ X, _Float16* __restrict__ Y,
                  int cols) {
  int row = blockIdx.x;
  const float* x = X + (size_t)row * cols;
  _Float16* y = Y + (size_t)row * cols;
  __shared__ float rs_[8], rq_[8];
  int tid = threadIdx.x, lane = tid & 31, wave = tid >> 5;
  float s = 0.f, q = 0.f;
  for (int c = tid; c < cols; c += 256) { float v = x[c]; s += v; q += v * v; }
  for (int o = 16; o > 0; o >>= 1) {
    s += __shfl_xor(s, o, 32);
    q += __shfl_xor(q, o, 32);
  }
  if (lane == 0) { rs_[wave] = s; rq_[wave] = q; }
  __syncthreads();
  s = 0.f; q = 0.f;
  for (int i = 0; i < 8; i++) { s += rs_[i]; q += rq_[i]; }
  float m = s / cols;
  float r = rsqrtf(q / cols - m * m + 1e-5f);
  for (int c = tid; c < cols; c += 256) {
    float v = (x[c] - m) * r;
    y[c] = (_Float16)fmaxf(v, 0.f);
  }
}

// ---------------------------------------------------------------------------
// Generic WMMA GEMM: C[M,N] = A[M,K]*B[K,N] (+bias per row), f16 in, f32 out.
// Block tile 128x128, 8 waves each 32x64, K-step 32. blockIdx.z batches.
// Requires M % 128 == 0, K % 32 == 0.
// N handling: every launched shape has N == 2048 (all 16-half B chunks fully
// in-bounds) or N == 64 (each chunk fully in- or fully out-of-bounds, since
// chunk starts step by 16) -- so B staging is "full chunk or zeros", no
// partial-element path (avoids scalarized saveexec region in codegen).
// ---------------------------------------------------------------------------
__global__ __launch_bounds__(256)
void k_gemm_f16(const _Float16* __restrict__ A, const _Float16* __restrict__ B,
                const float* __restrict__ bias, float* __restrict__ C,
                int M, int N, int K, size_t sA, size_t sB, size_t sC) {
  A += sA * blockIdx.z;
  B += sB * blockIdx.z;
  C += sC * blockIdx.z;
  alignas(16) __shared__ _Float16 As[128 * 40];   // row stride 40 halves (80B)
  alignas(16) __shared__ _Float16 Bt[128 * 40];   // [col][k], stride 40

  const int tid = threadIdx.x;
  const int lane = tid & 31, wave = tid >> 5;
  const int m0 = blockIdx.y * 128;
  const int n0 = blockIdx.x * 128;
  const int wm = (wave >> 1) * 32;
  const int wn = (wave & 1) * 64;
  v8f acc[2][4] = {};

  const int ar = tid & 127;            // A staging row
  const int apart = (tid >> 7) * 16;   // halves 0..15 or 16..31
  const int bkk = tid >> 3;            // B staging k row (0..31)
  const int bc0 = (tid & 7) * 16;      // B staging col chunk
  const bool bin = (n0 + bc0 + 16) <= N;  // chunk fully in-bounds?
  const int arow = lane & 15;
  const int ks0 = (lane < 16) ? 0 : 8;   // A frag K sub-offset per ISA layout
  const int bcol = lane & 15;
  const int kb = (lane < 16) ? 0 : 8;    // B frag K sub-offset per ISA layout

  for (int k0 = 0; k0 < K; k0 += 32) {
    __syncthreads();
    {  // stage A tile [128 x 32]
      const _Float16* ap = A + (size_t)(m0 + ar) * K + k0 + apart;
      *(v8h*)&As[ar * 40 + apart] = *(const v8h*)ap;
      *(v8h*)&As[ar * 40 + apart + 8] = *(const v8h*)(ap + 8);
      if (k0 + 32 < K) __builtin_prefetch(ap + 32, 0, 1);
    }
    {  // stage B tile [32 x 128] transposed -> Bt[col][k]
      const _Float16* bp = B + (size_t)(k0 + bkk) * N + n0 + bc0;
      v8h b0 = {}, b1 = {};
      if (bin) {
        b0 = *(const v8h*)bp;
        b1 = *(const v8h*)(bp + 8);
        if (k0 + 32 < K) __builtin_prefetch(bp + (size_t)32 * N, 0, 1);
      }
#pragma unroll
      for (int i = 0; i < 8; i++) Bt[(bc0 + i) * 40 + bkk] = b0[i];
#pragma unroll
      for (int i = 0; i < 8; i++) Bt[(bc0 + 8 + i) * 40 + bkk] = b1[i];
    }
    __syncthreads();
    v16h afrg[2], bfrg[4];
#pragma unroll
    for (int i = 0; i < 2; i++) {
      int r = wm + i * 16 + arow;
      afrg[i] = mkv16(*(const v8h*)&As[r * 40 + ks0],
                      *(const v8h*)&As[r * 40 + 16 + ks0]);
    }
#pragma unroll
    for (int j = 0; j < 4; j++) {
      int c = wn + j * 16 + bcol;
      bfrg[j] = mkv16(*(const v8h*)&Bt[c * 40 + kb],
                      *(const v8h*)&Bt[c * 40 + kb + 16]);
    }
#pragma unroll
    for (int i = 0; i < 2; i++)
#pragma unroll
      for (int j = 0; j < 4; j++)
        acc[i][j] = wmma16(afrg[i], bfrg[j], acc[i][j]);
  }

  const int crow0 = (lane < 16) ? 0 : 8;
  const int ccol = lane & 15;
#pragma unroll
  for (int i = 0; i < 2; i++)
#pragma unroll
    for (int j = 0; j < 4; j++) {
      int n = n0 + wn + j * 16 + ccol;
      if (n < N) {
        int mb = m0 + wm + i * 16 + crow0;
#pragma unroll
        for (int vr = 0; vr < 8; vr++) {
          float v = acc[i][j][vr];
          if (bias) v += bias[mb + vr];
          C[(size_t)(mb + vr) * N + n] = v;
        }
      }
    }
}

// ---------------------------------------------------------------------------
// Fused per-point PointNet: MLP 1->64->128->256->512 (IN+ReLU between),
// max-pool over K=64, final 512->256, L2 normalize.
// One block (256 threads) per point; 4096 points (src then tgt).
// ---------------------------------------------------------------------------
__device__ inline void inorm_to_curT(float* nxt, _Float16* curT, int Crows) {
  for (int c = threadIdx.x; c < Crows; c += 256) {
    float s = 0.f, q = 0.f;
    for (int col = 0; col < 64; col++) {
      float v = nxt[c * 66 + col];
      s += v; q += v * v;
    }
    float m = s * (1.0f / 64.0f);
    float r = rsqrtf(q * (1.0f / 64.0f) - m * m + 1e-5f);
    for (int col = 0; col < 64; col++) {
      float v = (nxt[c * 66 + col] - m) * r;
      curT[col * 264 + c] = (_Float16)fmaxf(v, 0.f);
    }
  }
}

__device__ inline void pn_gemm(const _Float16* __restrict__ W,
                               const float* __restrict__ bias, int Cout,
                               int Cin, const _Float16* curT, float* nxt,
                               float* g, bool dmax) {
  const int tid = threadIdx.x, lane = tid & 31, wave = tid >> 5;
  const int arow = lane & 15, ks0 = (lane < 16) ? 0 : 8;
  const int bcol = lane & 15, kb = (lane < 16) ? 0 : 8;
  const int ntiles = (Cout >> 4) * 4;   // m-tiles x 4 n-tiles (Ncols = 64)
  for (int t = wave; t < ntiles; t += 8) {
    const int mt = t >> 2, nt = t & 3;
    v8f acc = {};
    for (int k0 = 0; k0 < Cin; k0 += 32) {
      const _Float16* ap = W + (size_t)(mt * 16 + arow) * Cin + k0 + ks0;
      v16h a = mkv16(*(const v8h*)ap, *(const v8h*)(ap + 16));
      const _Float16* bp = curT + (nt * 16 + bcol) * 264 + k0 + kb;
      v16h b = mkv16(*(const v8h*)bp, *(const v8h*)(bp + 16));
      acc = wmma16(a, b, acc);
    }
    const int crow = mt * 16 + ((lane < 16) ? 0 : 8);
    const int ccol = nt * 16 + (lane & 15);
    if (!dmax) {
#pragma unroll
      for (int vr = 0; vr < 8; vr++)
        nxt[(crow + vr) * 66 + ccol] = acc[vr] + bias[crow + vr];
    } else {
#pragma unroll
      for (int vr = 0; vr < 8; vr++) {
        float v = acc[vr] + bias[crow + vr];
        for (int o = 8; o > 0; o >>= 1) v = fmaxf(v, __shfl_xor(v, o, 32));
        if ((lane & 15) == 0) atomicMaxF(&g[crow + vr], v);
      }
    }
  }
}

__global__ __launch_bounds__(256)
void k_pointnet(const float* __restrict__ src_geod,
                const float* __restrict__ tgt_geod,
                const _Float16* __restrict__ W1, const float* __restrict__ b1,
                const _Float16* __restrict__ W2, const float* __restrict__ b2,
                const _Float16* __restrict__ W3, const float* __restrict__ b3,
                const _Float16* __restrict__ W4, const float* __restrict__ b4,
                const _Float16* __restrict__ Wf, const float* __restrict__ bfin,
                float* __restrict__ pe_src, float* __restrict__ pe_tgt,
                float* __restrict__ out_src_pe,
                float* __restrict__ out_tgt_pe) {
  alignas(16) __shared__ _Float16 curT[64 * 264];  // [col][chan] transposed act
  __shared__ float nxt[256 * 66];                  // [chan][col]
  __shared__ float g[512];
  __shared__ float red[8];
  const int tid = threadIdx.x, lane = tid & 31, wave = tid >> 5;
  const int p = blockIdx.x;
  const float* x0 =
      (p < NTOK) ? (src_geod + (size_t)p * 64) : (tgt_geod + (size_t)(p - NTOK) * 64);

  // layer 1: 1 -> 64 (scalar outer product)
  for (int i = tid; i < 64 * 64; i += 256) {
    int c = i >> 6, col = i & 63;
    nxt[c * 66 + col] = (float)W1[c] * x0[col] + b1[c];
  }
  __syncthreads();
  inorm_to_curT(nxt, curT, 64);
  __syncthreads();
  pn_gemm(W2, b2, 128, 64, curT, nxt, nullptr, false);   // 64 -> 128
  __syncthreads();
  inorm_to_curT(nxt, curT, 128);
  __syncthreads();
  pn_gemm(W3, b3, 256, 128, curT, nxt, nullptr, false);  // 128 -> 256
  __syncthreads();
  inorm_to_curT(nxt, curT, 256);
  for (int i = tid; i < 512; i += 256) g[i] = -3.0e38f;
  __syncthreads();
  pn_gemm(W4, b4, 512, 256, curT, nullptr, g, true);     // 256 -> 512 + maxpool
  __syncthreads();

  // final: out[tid] = Wf[tid,:] . g + bfin[tid], then L2 normalize
  float ov = bfin[tid];
  const _Float16* wr = Wf + (size_t)tid * 512;
  for (int c2 = 0; c2 < 512; c2 += 8) {
    v8h wv = *(const v8h*)(wr + c2);
#pragma unroll
    for (int t = 0; t < 8; t++) ov += (float)wv[t] * g[c2 + t];
  }
  float ss = ov * ov;
  for (int o = 16; o > 0; o >>= 1) ss += __shfl_xor(ss, o, 32);
  if (lane == 0) red[wave] = ss;
  __syncthreads();
  ss = 0.f;
  for (int i = 0; i < 8; i++) ss += red[i];
  float inv = 1.f / fmaxf(sqrtf(ss), 1e-12f);
  float val = ov * inv;
  if (p < NTOK) {
    pe_src[(size_t)tid * NTOK + p] = val;
    out_src_pe[(size_t)p * 256 + tid] = val;
  } else {
    int pp = p - NTOK;
    pe_tgt[(size_t)tid * NTOK + pp] = val;
    out_tgt_pe[(size_t)pp * 256 + tid] = val;
  }
}

// ---------------------------------------------------------------------------
// Host orchestration
// ---------------------------------------------------------------------------
extern "C" void kernel_launch(void* const* d_in, const int* in_sizes, int n_in,
                              void* d_out, int out_size, void* d_ws,
                              size_t ws_size, hipStream_t stream) {
  (void)in_sizes; (void)out_size; (void)ws_size;
  const float* src_geod = (const float*)d_in[0];   // [2048,64]
  const float* tgt_geod = (const float*)d_in[1];
  const float* src_feats = (const float*)d_in[2];  // [2048,256]
  const float* tgt_feats = (const float*)d_in[3];

  // params: 84 leaves, jax pytree (sorted dict key) order:
  // final_proj{W,b}; layers[0..5]{merge,mlp1,mlp2,proj0..2}(W,b each);
  // pointnet{final(W,b), mlp0..3(W,b)}
  int leaf_sizes[84];
  leaf_sizes[0] = 65536; leaf_sizes[1] = 256;
  for (int li = 0; li < 6; li++) {
    int b = 2 + 12 * li;
    leaf_sizes[b + 0] = 65536;  leaf_sizes[b + 1] = 256;   // merge
    leaf_sizes[b + 2] = 262144; leaf_sizes[b + 3] = 512;   // mlp1
    leaf_sizes[b + 4] = 131072; leaf_sizes[b + 5] = 256;   // mlp2
    leaf_sizes[b + 6] = 65536;  leaf_sizes[b + 7] = 256;   // proj q
    leaf_sizes[b + 8] = 65536;  leaf_sizes[b + 9] = 256;   // proj k
    leaf_sizes[b + 10] = 65536; leaf_sizes[b + 11] = 256;  // proj v
  }
  leaf_sizes[74] = 131072; leaf_sizes[75] = 256;  // pointnet final
  leaf_sizes[76] = 64;     leaf_sizes[77] = 64;   // mlp0
  leaf_sizes[78] = 8192;   leaf_sizes[79] = 128;  // mlp1
  leaf_sizes[80] = 32768;  leaf_sizes[81] = 256;  // mlp2
  leaf_sizes[82] = 131072; leaf_sizes[83] = 512;  // mlp3

  const float* P[84];
  if (n_in >= 4 + 84) {
    for (int i = 0; i < 84; i++) P[i] = (const float*)d_in[4 + i];
  } else {
    const float* base = (const float*)d_in[4];
    size_t o = 0;
    for (int i = 0; i < 84; i++) { P[i] = base + o; o += (size_t)leaf_sizes[i]; }
  }

  // workspace bump allocator
  char* wsb = (char*)d_ws;
  size_t woff = 0;
  auto alloc = [&](size_t bytes) -> void* {
    void* p = wsb + woff;
    woff = (woff + bytes + 255) & ~(size_t)255;
    return p;
  };
  _Float16* w16 = (_Float16*)alloc((size_t)4300864 * 2);  // f16 weight pool
  float* pe_src = (float*)alloc((size_t)524288 * 4);
  float* pe_tgt = (float*)alloc((size_t)524288 * 4);
  float* S = (float*)alloc((size_t)524288 * 4);
  float* T = (float*)alloc((size_t)524288 * 4);
  float* Sold = (float*)alloc((size_t)524288 * 4);
  float* qf = (float*)alloc((size_t)524288 * 4);
  float* kf = (float*)alloc((size_t)524288 * 4);
  float* vf = (float*)alloc((size_t)524288 * 4);
  float* msg = (float*)alloc((size_t)524288 * 4);
  float* dlt = (float*)alloc((size_t)524288 * 4);
  float* zf = (float*)alloc((size_t)1048576 * 4);
  float* attn = (float*)alloc((size_t)524288 * 4);
  float* Cf = (float*)alloc((size_t)524288 * 4);
  float* scores = (float*)alloc((size_t)16777216 * 4);
  _Float16* s16 = (_Float16*)alloc((size_t)524288 * 2);
  _Float16* y16 = (_Float16*)alloc((size_t)1048576 * 2);  // rows 0-255 x, 256-511 msg
  _Float16* z16 = (_Float16*)alloc((size_t)1048576 * 2);
  _Float16* xo16 = (_Float16*)alloc((size_t)524288 * 2);
  _Float16* qT16 = (_Float16*)alloc((size_t)524288 * 2);
  _Float16* k16 = (_Float16*)alloc((size_t)524288 * 2);
  _Float16* vT16 = (_Float16*)alloc((size_t)524288 * 2);
  _Float16* prob16 = (_Float16*)alloc((size_t)16777216 * 2);

  auto conv = [&](const float* src, _Float16* dst, int n) {
    k_f32_to_f16<<<(n + 255) / 256, 256, 0, stream>>>(dst, src, n);
  };
  const size_t OFF_FP = 0;
  auto Lw = [](int li) -> size_t { return 65536 + (size_t)li * 655360; };
  const size_t PNB = 65536 + (size_t)6 * 655360;

  // convert all GEMM weights to f16
  conv(P[0], w16 + OFF_FP, 65536);
  for (int li = 0; li < 6; li++) {
    int b = 2 + 12 * li; size_t L = Lw(li);
    conv(P[b + 0], w16 + L + 0, 65536);        // merge
    conv(P[b + 2], w16 + L + 65536, 262144);   // mlp1
    conv(P[b + 4], w16 + L + 327680, 131072);  // mlp2
    conv(P[b + 6], w16 + L + 458752, 65536);   // q
    conv(P[b + 8], w16 + L + 524288, 65536);   // k
    conv(P[b + 10], w16 + L + 589824, 65536);  // v
  }
  conv(P[74], w16 + PNB + 0, 131072);       // Wf
  conv(P[76], w16 + PNB + 131072, 64);      // W1
  conv(P[78], w16 + PNB + 131136, 8192);    // W2
  conv(P[80], w16 + PNB + 139328, 32768);   // W3
  conv(P[82], w16 + PNB + 172096, 131072);  // W4

  float* out_f = (float*)d_out;  // [src | tgt | src_pe | tgt_pe], each 524288

  // PointNet positional embeddings (fills pe_* and out sections 2,3)
  k_pointnet<<<4096, 256, 0, stream>>>(
      src_geod, tgt_geod, w16 + PNB + 131072, P[77], w16 + PNB + 131136, P[79],
      w16 + PNB + 139328, P[81], w16 + PNB + 172096, P[83], w16 + PNB + 0,
      P[75], pe_src, pe_tgt, out_f + (size_t)2 * 524288,
      out_f + (size_t)3 * 524288);

  k_transpose_in<<<2048, 256, 0, stream>>>(S, src_feats);
  k_transpose_in<<<2048, 256, 0, stream>>>(T, tgt_feats);

  auto gemmz = [&](const _Float16* A, const _Float16* B, const float* bias,
                   float* C, int M, int N, int K, int Z, size_t sA, size_t sB,
                   size_t sC) {
    dim3 grid((N + 127) / 128, M / 128, Z);
    k_gemm_f16<<<grid, 256, 0, stream>>>(A, B, bias, C, M, N, K, sA, sB, sC);
  };
  auto gemm = [&](const _Float16* A, const _Float16* B, const float* bias,
                  float* C, int M, int N, int K) {
    gemmz(A, B, bias, C, M, N, K, 1, 0, 0, 0);
  };
  auto addv = [&](float* a, const float* b) {
    k_add_inplace<<<2048, 256, 0, stream>>>(a, b, 524288);
  };

  // one cross-attention block; result delta left in dlt
  auto cross_attn = [&](const float* X, const float* SRC, int li) {
    size_t L = Lw(li); int b = 2 + 12 * li;
    k_f32_to_f16<<<2048, 256, 0, stream>>>(y16, X, 524288);        // x16
    k_f32_to_f16<<<2048, 256, 0, stream>>>(s16, SRC, 524288);
    gemm(w16 + L + 458752, y16, P[b + 7], qf, 256, 2048, 256);     // q
    gemm(w16 + L + 524288, s16, P[b + 9], kf, 256, 2048, 256);     // k
    gemm(w16 + L + 589824, s16, P[b + 11], vf, 256, 2048, 256);    // v
    k_qkv_pack<<<2048, 256, 0, stream>>>(qf, kf, vf, qT16, k16, vT16);
    // scores_h = qT_h[2048x64] * k_h[64x2048], 4 heads via blockIdx.z
    gemmz(qT16, k16, nullptr, scores, 2048, 2048, 64, 4, 131072, 131072,
          4194304);
    k_softmax<<<8192, 256, 0, stream>>>(scores, prob16, 2048, 0.125f);
    // attn_h = prob_h[2048x2048] * vT_h[2048x64]
    gemmz(prob16, vT16, nullptr, attn, 2048, 64, 2048, 4, 4194304, 131072,
          131072);
    k_attn_pack<<<2048, 256, 0, stream>>>(attn, xo16);
    gemm(w16 + L + 0, xo16, P[b + 1], msg, 256, 2048, 256);        // merge
    k_f32_to_f16<<<2048, 256, 0, stream>>>(y16 + 524288, msg, 524288);
    gemm(w16 + L + 65536, y16, P[b + 3], zf, 512, 2048, 512);      // mlp1
    k_inorm_relu<<<512, 256, 0, stream>>>(zf, z16, 2048);
    gemm(w16 + L + 327680, z16, P[b + 5], dlt, 256, 2048, 512);    // mlp2
  };

  const int arch_self[6] = {1, 0, 1, 0, 1, 0};
  for (int li = 0; li < 6; li++) {
    if (arch_self[li]) {
      addv(S, pe_src);
      addv(T, pe_tgt);
      cross_attn(S, S, li); addv(S, dlt);
      cross_attn(T, T, li); addv(T, dlt);
    } else {
      hipMemcpyAsync(Sold, S, (size_t)524288 * 4, hipMemcpyDeviceToDevice,
                     stream);
      cross_attn(S, T, li); addv(S, dlt);
      cross_attn(T, Sold, li); addv(T, dlt);
    }
  }

  // final projection + transpose out
  k_f32_to_f16<<<2048, 256, 0, stream>>>(s16, S, 524288);
  gemm(w16 + OFF_FP, s16, P[1], Cf, 256, 2048, 256);
  k_transpose_out<<<2048, 256, 0, stream>>>(out_f + 0, Cf);
  k_f32_to_f16<<<2048, 256, 0, stream>>>(s16, T, 524288);
  gemm(w16 + OFF_FP, s16, P[1], Cf, 256, 2048, 256);
  k_transpose_out<<<2048, 256, 0, stream>>>(out_f + 524288, Cf);
}
